// MultiFlowClassifierRouter_62878321214036
// MI455X (gfx1250) — matrix operand order repaired and checked by reading, not compiled, not run on hardware
//
#include <hip/hip_runtime.h>
#include <hip/hip_bf16.h>

// Problem sizes (fixed by the reference)
#define M_EXP 8
#define B_N   4096
#define DIN   1024
#define DHID  4096
#define DOUT  1024

// Block tile: 128 rows x 128 cols, 256 threads = 8 waves.
// Wave w: row-group rg = w>>1 (32 rows), col-group cg = w&1 (64 cols)
//  -> 2 A-frags x 4 B-frags = 8 WMMA accumulators per wave per K-slab (K=32).

typedef __attribute__((ext_vector_type(16))) __bf16 v16bf;
typedef __attribute__((ext_vector_type(2)))  __bf16 v2bf;
typedef __attribute__((ext_vector_type(2)))  float  v2f;
typedef __attribute__((ext_vector_type(8)))  float  v8f;
typedef __attribute__((ext_vector_type(4)))  int    v4i;

// ---- gfx1250 async global->LDS (guarded) ---------------------------------
#if __has_builtin(__builtin_amdgcn_global_load_async_to_lds_b128)
#define HAVE_ASYNC_LDS 1
__device__ __forceinline__ void async_copy_b128(const void* g, void* l) {
    __builtin_amdgcn_global_load_async_to_lds_b128(
        (__attribute__((address_space(1))) v4i*)g,
        (__attribute__((address_space(3))) v4i*)l, 0, 0);
}
#else
#define HAVE_ASYNC_LDS 0
#endif

#if __has_builtin(__builtin_amdgcn_s_wait_asynccnt)
#define WAIT_ASYNC(n) __builtin_amdgcn_s_wait_asynccnt(n)
#else
#define WAIT_ASYNC(n) asm volatile("s_wait_asynccnt %0" :: "i"(n) : "memory")
#endif

// ---- helpers -------------------------------------------------------------

__device__ __forceinline__ unsigned short f2bf(float f) {
    unsigned u = __builtin_bit_cast(unsigned, f);
    u += 0x7fffu + ((u >> 16) & 1u);
    return (unsigned short)(u >> 16);
}

// pack two f32 -> one dword of 2 bf16 (RNE); prefer a HW packed convert
__device__ __forceinline__ unsigned pack_bf2(float a, float b) {
#if __has_builtin(__builtin_amdgcn_cvt_pk_bf16_f32)
    v2bf r = __builtin_amdgcn_cvt_pk_bf16_f32(a, b);
    return __builtin_bit_cast(unsigned, r);
#else
    v2f f = {a, b};
    v2bf r = __builtin_convertvector(f, v2bf);
    return __builtin_bit_cast(unsigned, r);
#endif
}

__device__ __forceinline__ float gelu_tanh(float x) {
    const float c0 = 0.7978845608028654f;   // sqrt(2/pi)
    const float c1 = 0.044715f;
    float t = tanhf(c0 * (x + c1 * x * x * x));
    return 0.5f * x * (1.0f + t);
}

union FragU { uint4 q[2]; v16bf v; };

// A fragment (LDS row-major [row][32K], stride 32):
// lane<16 -> row=lane, K {b..b+7} U {b+16..b+23}, b=(lane>>4)*8
__device__ __forceinline__ v16bf load_frag_a(const unsigned short* p) {
    FragU u;
    u.q[0] = *(const uint4*)(p);
    u.q[1] = *(const uint4*)(p + 16);
    return u.v;
}
// B fragment (LDS transposed [N][K], stride 40): K {b..b+7} U {b+16..b+23}, b=(lane>>4)*8
__device__ __forceinline__ v16bf load_frag_b(const unsigned short* p) {
    FragU u;
    u.q[0] = *(const uint4*)(p);
    u.q[1] = *(const uint4*)(p + 16);
    return u.v;
}

// ---- routing kernels -----------------------------------------------------

__global__ void route_zero_kernel(int* counts) {
    if (threadIdx.x < 64) counts[threadIdx.x] = 0;
}

__global__ void route_count_kernel(const long long* __restrict__ route,
                                   int* __restrict__ counts) {
    int i = blockIdx.x * 256 + threadIdx.x;
    if (i < B_N) atomicAdd(&counts[(int)route[i]], 1);
}

__global__ void route_scan_kernel(const int* __restrict__ counts,
                                  int* __restrict__ offsets) {
    if (threadIdx.x == 0) {
        int s = 0;
        for (int m = 0; m < M_EXP; ++m) { offsets[m] = s; s += counts[m]; }
        offsets[M_EXP] = s;
    }
}

__global__ void route_scatter_kernel(const long long* __restrict__ route,
                                     const int* __restrict__ offsets,
                                     int* __restrict__ cursors,
                                     int* __restrict__ idx_map) {
    int i = blockIdx.x * 256 + threadIdx.x;
    if (i < B_N) {
        int e   = (int)route[i];
        int pos = atomicAdd(&cursors[e], 1);
        idx_map[offsets[e] + pos] = i;
    }
}

// ---- GEMM1: H = gelu(Xg @ W1[e] + b1[e]) (bf16 out) ----------------------
// grid = (DHID/128, M_EXP * B_N/128), block = 256

__global__ __launch_bounds__(256) void gemm1_kernel(
    const float* __restrict__ x, const float* __restrict__ W1,
    const float* __restrict__ b1, const int* __restrict__ counts,
    const int* __restrict__ offsets, const int* __restrict__ idx_map,
    unsigned short* __restrict__ H)
{
    const int e   = blockIdx.y >> 5;
    const int rt  = blockIdx.y & 31;
    const int cnt = counts[e];
    if (rt * 128 >= cnt) return;
    const int base = offsets[e];
    const int n0   = blockIdx.x * 128;

    __shared__ __align__(16) unsigned short sA[128 * 32];   // [row][K]
    __shared__ __align__(16) unsigned short sB[128 * 40];   // [N][K], pad 40
    __shared__ int srow[128];

    const int tid  = threadIdx.x;
    const int wave = tid >> 5;
    const int lane = tid & 31;
    const int rg   = wave >> 1;
    const int cg   = wave & 1;
    const int l15  = lane & 15;
    const int lhi  = lane >> 4;

    if (tid < 128) {
        int r = rt * 128 + tid;
        srow[tid] = (r < cnt) ? idx_map[base + r] : -1;
    }
    __syncthreads();

    const unsigned long long wbase = ((unsigned long long)e) << 22; // e*DIN*DHID

    // --- per-thread staging pointers (advance by fixed stride per K-slab) --
    // A: 4 units, each float4 (row r, K-quad kq). Clamp padded rows: a garbage
    //    A row only pollutes the same (discarded) C row.
    const float* gA[4]; unsigned short* lA[4];
#pragma unroll
    for (int i = 0; i < 4; ++i) {
        int u = tid + i * 256;
        int r = u >> 3, kq = (u & 7) * 4;
        int row = srow[r];
        int rowc = row < 0 ? 0 : row;
        gA[i] = x + (unsigned)rowc * DIN + kq;
        lA[i] = &sA[r * 32 + kq];
    }
    // B: 4 units, each 2 cols x 2 K-rows (two float2 loads, two b32 stores)
    const float* gB[4]; unsigned short* lB[4];
#pragma unroll
    for (int i = 0; i < 4; ++i) {
        int u = tid + i * 256;
        int nn2 = (u & 63) * 2;
        int kk  = (u >> 6) * 2;
        gB[i] = W1 + wbase + (unsigned long long)kk * DHID + (n0 + nn2);
        lB[i] = &sB[nn2 * 40 + kk];
    }
    // fragment read pointers (constant across iterations)
    const unsigned short* fA0 = &sA[(rg * 32 +      l15) * 32 + lhi * 8];
    const unsigned short* fA1 = &sA[(rg * 32 + 16 + l15) * 32 + lhi * 8];
    const unsigned short* fB[4];
#pragma unroll
    for (int j = 0; j < 4; ++j)
        fB[j] = &sB[(cg * 64 + j * 16 + l15) * 40 + lhi * 8];

    // --- software pipeline: regs(k) -> LDS while loading regs(k+1) ---------
    float4 ra[4]; v2f rb0[4], rb1[4];
#pragma unroll
    for (int i = 0; i < 4; ++i) ra[i] = *(const float4*)gA[i];
#pragma unroll
    for (int i = 0; i < 4; ++i) {
        rb0[i] = *(const v2f*)gB[i];
        rb1[i] = *(const v2f*)(gB[i] + DHID);
    }

    v8f acc[2][4] = {};
    const int NK = DIN / 32;
    for (int kt = 0; kt < NK; ++kt) {
        __syncthreads();                       // frag reads of kt-1 complete
#pragma unroll
        for (int i = 0; i < 4; ++i) {
            uint2 p = make_uint2(pack_bf2(ra[i].x, ra[i].y),
                                 pack_bf2(ra[i].z, ra[i].w));
            *(uint2*)lA[i] = p;
        }
#pragma unroll
        for (int i = 0; i < 4; ++i) {
            *(unsigned*)(lB[i])      = pack_bf2(rb0[i].x, rb1[i].x);
            *(unsigned*)(lB[i] + 40) = pack_bf2(rb0[i].y, rb1[i].y);
        }
        if (kt + 1 < NK) {                     // issue next K-slab's loads
#pragma unroll
            for (int i = 0; i < 4; ++i) { gA[i] += 32; ra[i] = *(const float4*)gA[i]; }
#pragma unroll
            for (int i = 0; i < 4; ++i) {
                gB[i] += 32 * DHID;
                rb0[i] = *(const v2f*)gB[i];
                rb1[i] = *(const v2f*)(gB[i] + DHID);
            }
            __builtin_prefetch(gB[0] + 32 * DHID, 0, 0);   // global_prefetch_b8
        }
        __syncthreads();                       // staging visible to all waves
        v16bf a0 = load_frag_a(fA0);
        v16bf a1 = load_frag_a(fA1);
#pragma unroll
        for (int j = 0; j < 4; ++j) {
            v16bf b = load_frag_b(fB[j]);
            acc[0][j] = __builtin_amdgcn_wmma_f32_16x16x32_bf16(false, a0, false, b,
                                                                (short)0, acc[0][j], false, false);
            acc[1][j] = __builtin_amdgcn_wmma_f32_16x16x32_bf16(false, a1, false, b,
                                                                (short)0, acc[1][j], false, false);
        }
    }

    // epilogue: bias + gelu -> H (bf16, compact rows)
#pragma unroll
    for (int j = 0; j < 4; ++j) {
        const int colg = n0 + cg * 64 + j * 16 + l15;
        const float bias = b1[e * DHID + colg];
#pragma unroll
        for (int ri = 0; ri < 2; ++ri) {
#pragma unroll
            for (int i = 0; i < 8; ++i) {
                int rloc = rg * 32 + ri * 16 + i + lhi * 8;   // C/D row layout
                int r = rt * 128 + rloc;
                if (r < cnt) {
                    float hv = gelu_tanh(acc[ri][j][i] + bias);
                    H[(unsigned long long)(base + r) * DHID + colg] = f2bf(hv);
                }
            }
        }
    }
}

// ---- GEMM2: out = H @ W2[e] + b2[e], scattered to sample rows ------------
// grid = (DOUT/128, M_EXP * B_N/128), block = 256
// A tile is already bf16 -> copy with gfx1250 async global->LDS (double buffer)

__global__ __launch_bounds__(256) void gemm2_kernel(
    const unsigned short* __restrict__ H, const float* __restrict__ W2,
    const float* __restrict__ b2, const int* __restrict__ counts,
    const int* __restrict__ offsets, const int* __restrict__ idx_map,
    float* __restrict__ out)
{
    const int e   = blockIdx.y >> 5;
    const int rt  = blockIdx.y & 31;
    const int cnt = counts[e];
    if (rt * 128 >= cnt) return;
    const int base = offsets[e];
    const int n0   = blockIdx.x * 128;

    __shared__ __align__(16) unsigned short sA2[2][128 * 32];  // double buffer
    __shared__ __align__(16) unsigned short sB[128 * 40];
    __shared__ int srow[128];

    const int tid  = threadIdx.x;
    const int wave = tid >> 5;
    const int lane = tid & 31;
    const int rg   = wave >> 1;
    const int cg   = wave & 1;
    const int l15  = lane & 15;
    const int lhi  = lane >> 4;

    if (tid < 128) {
        int r = rt * 128 + tid;
        srow[tid] = (r < cnt) ? idx_map[base + r] : -1;
    }
    __syncthreads();

    const unsigned long long wbase = ((unsigned long long)e) << 22; // e*DHID*DOUT

    // A: 2 units/thread, 8 bf16 (b128) each; clamp rows past cnt (results discarded)
    const unsigned short* gH[2]; int lAoff[2];
#pragma unroll
    for (int i = 0; i < 2; ++i) {
        int u = tid + i * 256;
        int r = u >> 2, k8 = (u & 3) * 8;
        int rr = rt * 128 + r;
        int rrc = rr < cnt ? rr : cnt - 1;
        gH[i]   = H + (unsigned long long)(base + rrc) * DHID + k8;
        lAoff[i] = r * 32 + k8;
    }
    const float* gB[4]; unsigned short* lB[4];
#pragma unroll
    for (int i = 0; i < 4; ++i) {
        int u = tid + i * 256;
        int nn2 = (u & 63) * 2;
        int kk  = (u >> 6) * 2;
        gB[i] = W2 + wbase + (unsigned long long)kk * DOUT + (n0 + nn2);
        lB[i] = &sB[nn2 * 40 + kk];
    }
    const unsigned short* fB[4];
#pragma unroll
    for (int j = 0; j < 4; ++j)
        fB[j] = &sB[(cg * 64 + j * 16 + l15) * 40 + lhi * 8];

    // pipeline priming
#if HAVE_ASYNC_LDS
#pragma unroll
    for (int i = 0; i < 2; ++i)
        async_copy_b128(gH[i], &sA2[0][lAoff[i]]);
#else
    uint4 rha[2];
#pragma unroll
    for (int i = 0; i < 2; ++i) rha[i] = *(const uint4*)gH[i];
#endif
    v2f rb0[4], rb1[4];
#pragma unroll
    for (int i = 0; i < 4; ++i) {
        rb0[i] = *(const v2f*)gB[i];
        rb1[i] = *(const v2f*)(gB[i] + DOUT);
    }

    v8f acc[2][4] = {};
    const int NK = DHID / 32;
    for (int kt = 0; kt < NK; ++kt) {
        const int buf = kt & 1;
        __syncthreads();
#if !HAVE_ASYNC_LDS
#pragma unroll
        for (int i = 0; i < 2; ++i) *(uint4*)&sA2[buf][lAoff[i]] = rha[i];
#endif
#pragma unroll
        for (int i = 0; i < 4; ++i) {
            *(unsigned*)(lB[i])      = pack_bf2(rb0[i].x, rb1[i].x);
            *(unsigned*)(lB[i] + 40) = pack_bf2(rb0[i].y, rb1[i].y);
        }
        if (kt + 1 < NK) {
#pragma unroll
            for (int i = 0; i < 2; ++i) gH[i] += 32;
#if HAVE_ASYNC_LDS
#pragma unroll
            for (int i = 0; i < 2; ++i)
                async_copy_b128(gH[i], &sA2[buf ^ 1][lAoff[i]]);
#else
#pragma unroll
            for (int i = 0; i < 2; ++i) rha[i] = *(const uint4*)gH[i];
#endif
#pragma unroll
            for (int i = 0; i < 4; ++i) {
                gB[i] += 32 * DOUT;
                rb0[i] = *(const v2f*)gB[i];
                rb1[i] = *(const v2f*)(gB[i] + DOUT);
            }
            __builtin_prefetch(gB[0] + 32 * DOUT, 0, 0);
#if HAVE_ASYNC_LDS
            WAIT_ASYNC(2);     // current buffer's 2 copies done; next 2 in flight
#endif
        } else {
#if HAVE_ASYNC_LDS
            WAIT_ASYNC(0);
#endif
        }
        __syncthreads();
        v16bf a0 = load_frag_a(&sA2[buf][(rg * 32 +      l15) * 32 + lhi * 8]);
        v16bf a1 = load_frag_a(&sA2[buf][(rg * 32 + 16 + l15) * 32 + lhi * 8]);
#pragma unroll
        for (int j = 0; j < 4; ++j) {
            v16bf b = load_frag_b(fB[j]);
            acc[0][j] = __builtin_amdgcn_wmma_f32_16x16x32_bf16(false, a0, false, b,
                                                                (short)0, acc[0][j], false, false);
            acc[1][j] = __builtin_amdgcn_wmma_f32_16x16x32_bf16(false, a1, false, b,
                                                                (short)0, acc[1][j], false, false);
        }
    }

    // epilogue: bias, scatter to original sample rows (f32)
#pragma unroll
    for (int j = 0; j < 4; ++j) {
        const int colg = n0 + cg * 64 + j * 16 + l15;
        const float bias = b2[e * DOUT + colg];
#pragma unroll
        for (int ri = 0; ri < 2; ++ri) {
#pragma unroll
            for (int i = 0; i < 8; ++i) {
                int rloc = rg * 32 + ri * 16 + i + lhi * 8;
                int r = rt * 128 + rloc;
                if (r < cnt) {
                    int sample = srow[rloc];
                    out[(unsigned long long)sample * DOUT + colg] = acc[ri][j][i] + bias;
                }
            }
        }
    }
}

// ---- launch --------------------------------------------------------------

extern "C" void kernel_launch(void* const* d_in, const int* in_sizes, int n_in,
                              void* d_out, int out_size, void* d_ws, size_t ws_size,
                              hipStream_t stream) {
    const float*     x     = (const float*)d_in[0];
    const long long* route = (const long long*)d_in[1];
    const float*     W1    = (const float*)d_in[2];
    const float*     b1    = (const float*)d_in[3];
    const float*     W2    = (const float*)d_in[4];
    const float*     b2    = (const float*)d_in[5];
    float*           out   = (float*)d_out;

    int* counts  = (int*)d_ws;            // 8 ints
    int* offsets = counts + 8;            // 9 ints
    int* cursors = counts + 32;           // 8 ints
    int* idx_map = counts + 64;           // B_N ints
    unsigned short* H = (unsigned short*)((char*)d_ws + 256 + B_N * sizeof(int));

    route_zero_kernel   <<<1, 64, 0, stream>>>(counts);
    route_count_kernel  <<<B_N / 256, 256, 0, stream>>>(route, counts);
    route_scan_kernel   <<<1, 32, 0, stream>>>(counts, offsets);
    route_scatter_kernel<<<B_N / 256, 256, 0, stream>>>(route, offsets, cursors, idx_map);

    dim3 g1(DHID / 128, M_EXP * (B_N / 128));
    gemm1_kernel<<<g1, 256, 0, stream>>>(x, W1, b1, counts, offsets, idx_map, H);

    dim3 g2(DOUT / 128, M_EXP * (B_N / 128));
    gemm2_kernel<<<g2, 256, 0, stream>>>(H, W2, b2, counts, offsets, idx_map, out);
}
